// LocalTransformer_64854006169855
// MI455X (gfx1250) — matrix operand order repaired and checked by reading, not compiled, not run on hardware
//
#include <hip/hip_runtime.h>
#include <hip/hip_bf16.h>
#include <cmath>

// ---------------------------------------------------------------------------
// Model constants (must match the reference)
// ---------------------------------------------------------------------------
#define DMODEL 512
#define NHEAD  8
#define HDIM   64
#define FFDIM  2048
#define NENC   4
#define NDEC   4
#define VOCAB  16000
#define BB     4
#define SS     1024
#define TTLEN  1024
#define WINSZ  128
#define NEGF   (-1.0e30f)
#define SCALEQ 0.125f           // 1/sqrt(HDIM)
#define SQRTD  22.62741699796952f

typedef __attribute__((ext_vector_type(16))) _Float16 v16h;
typedef __attribute__((ext_vector_type(8)))  _Float16 v8h;
typedef __attribute__((ext_vector_type(8)))  float    v8f;

// ---------------------------------------------------------------------------
// WMMA fragment loaders (wave32, V_WMMA_F32_16X16X32_F16 layouts; A/B use this file's contiguous-K order, not the ISA table order)
//
// A (16x32, MxK): lane L holds row m = L&15.  For L<16 the 16 halves map to
//   K = 0..15; for L>=16 to K = 16..31.  -> two 16B loads.
// B (32x16, KxN): lane L holds column n = L&15; halves are contiguous K,
//   lanes 0-15 -> K 0..15, lanes 16-31 -> K 16..31.  -> one 32B load, which
//   is why weights are stored transposed as Wt[N][K].
// C/D (16x16 f32): element r of the v8f in lane L is (m = r + 8*(L>>4),
//   n = L&15).
// ---------------------------------------------------------------------------
__device__ __forceinline__ v16h load_frag_a(const _Float16* A, int lda, int m0, int k0) {
    int lane = threadIdx.x & 31;
    int m  = m0 + (lane & 15);
    int hi = lane >> 4;
    const _Float16* p = A + (size_t)m * lda + k0 + hi * 16;
    v8h lo = *(const v8h*)(p);
    v8h hh = *(const v8h*)(p + 8);
    return __builtin_shufflevector(lo, hh, 0,1,2,3,4,5,6,7,8,9,10,11,12,13,14,15);
}

__device__ __forceinline__ v16h load_frag_b(const _Float16* Bt, int ldb, int n0, int k0) {
    int lane = threadIdx.x & 31;
    int n  = n0 + (lane & 15);
    int hi = lane >> 4;
    return *(const v16h*)(Bt + (size_t)n * ldb + k0 + hi * 16);
}

__device__ __forceinline__ v8f wmma_f16(v16h a, v16h b, v8f c) {
    return __builtin_amdgcn_wmma_f32_16x16x32_f16(false, a, false, b, (short)0, c, false, false);
}

// ---------------------------------------------------------------------------
// Generic GEMM:  C[M,N] = act( A[M,K] * Wt[N,K]^T + bias[N] )
// Block: 256 threads (8 waves), tile 128x128, each wave 32x64.
// All dims used here are multiples of 128 / 32 -> no bounds checks.
// ---------------------------------------------------------------------------
template<typename OutT, bool RELU>
__global__ __launch_bounds__(256) void gemm_wmma(
        const _Float16* __restrict__ A, const _Float16* __restrict__ Wt,
        const float* __restrict__ bias, OutT* __restrict__ C,
        int M, int N, int K) {
    const int wid  = threadIdx.x >> 5;
    const int lane = threadIdx.x & 31;
    const int m0 = blockIdx.y * 128 + (wid & 3) * 32;
    const int n0 = blockIdx.x * 128 + (wid >> 2) * 64;

    v8f acc[2][4];
#pragma unroll
    for (int i = 0; i < 2; ++i)
#pragma unroll
        for (int j = 0; j < 4; ++j) acc[i][j] = v8f{};

    for (int k0 = 0; k0 < K; k0 += 32) {
        v16h a0 = load_frag_a(A, K, m0,      k0);
        v16h a1 = load_frag_a(A, K, m0 + 16, k0);
#pragma unroll
        for (int j = 0; j < 4; ++j) {
            v16h b = load_frag_b(Wt, K, n0 + 16 * j, k0);
            acc[0][j] = wmma_f16(a0, b, acc[0][j]);
            acc[1][j] = wmma_f16(a1, b, acc[1][j]);
        }
    }

    const int hi = lane >> 4, nn = lane & 15;
#pragma unroll
    for (int i = 0; i < 2; ++i)
#pragma unroll
        for (int j = 0; j < 4; ++j) {
            int n = n0 + 16 * j + nn;
            float bv = bias[n];
#pragma unroll
            for (int r = 0; r < 8; ++r) {
                int m = m0 + 16 * i + r + hi * 8;
                float v = acc[i][j][r] + bv;
                if (RELU) v = fmaxf(v, 0.0f);
                C[(size_t)m * N + n] = (OutT)v;
            }
        }
}

// ---------------------------------------------------------------------------
// Fused flash attention.  1 wave / block; block handles one (b, h, 16-row
// q-tile).  mode: 0 = local window |i-j|<=WIN, 1 = causal window 0<=i-j<=WIN,
// 2 = full.  Q,K are [B*L, D] f16 (head h = cols h*64..); Vt is [B,H,HD,Lk].
// ---------------------------------------------------------------------------
__global__ __launch_bounds__(32) void attn_wmma(
        const _Float16* __restrict__ Q, const _Float16* __restrict__ Kc,
        const _Float16* __restrict__ Vt, const unsigned char* __restrict__ kpm,
        _Float16* __restrict__ O, int Lq, int Lk, int mode) {
    __shared__ _Float16 P[16 * 40];   // 16x32 P tile, stride 40 keeps 16B align
    const int lane = threadIdx.x & 31;
    const int hi = lane >> 4, nn = lane & 15;
    const int q0 = blockIdx.x * 16;
    const int h  = blockIdx.y;
    const int b  = blockIdx.z;

    const _Float16* Qh = Q  + (size_t)b * Lq * DMODEL + h * HDIM;
    const _Float16* Kh = Kc + (size_t)b * Lk * DMODEL + h * HDIM;
    const _Float16* Vh = Vt + (size_t)(b * NHEAD + h) * HDIM * Lk;
    const unsigned char* kp = kpm + (size_t)b * Lk;

    v16h qa0 = load_frag_a(Qh, DMODEL, q0, 0);
    v16h qa1 = load_frag_a(Qh, DMODEL, q0, 32);

    v8f oacc[4];
#pragma unroll
    for (int j = 0; j < 4; ++j) oacc[j] = v8f{};
    float mrun[8], lrun[8];
#pragma unroll
    for (int r = 0; r < 8; ++r) { mrun[r] = NEGF; lrun[r] = 0.0f; }

    int klo, khi;
    if (mode == 0)      { klo = max(q0 - WINSZ, 0); khi = min(Lk, q0 + 16 + WINSZ); }
    else if (mode == 1) { klo = max(q0 - WINSZ, 0); khi = min(Lk, q0 + 16); }
    else                { klo = 0;                  khi = Lk; }

    for (int kt = klo & ~31; kt < khi; kt += 32) {
        // scores for 32 keys: two 16x16 C tiles, HD=64 -> 2 WMMAs each
        v8f s0 = v8f{}, s1 = v8f{};
        s0 = wmma_f16(qa0, load_frag_b(Kh, DMODEL, kt,      0),  s0);
        s0 = wmma_f16(qa1, load_frag_b(Kh, DMODEL, kt,      32), s0);
        s1 = wmma_f16(qa0, load_frag_b(Kh, DMODEL, kt + 16, 0),  s1);
        s1 = wmma_f16(qa1, load_frag_b(Kh, DMODEL, kt + 16, 32), s1);

        const int k0i = kt + nn, k1i = kt + 16 + nn;
        const bool kp0 = kp[k0i] != 0;
        const bool kp1 = kp[k1i] != 0;

#pragma unroll
        for (int r = 0; r < 8; ++r) {
            int m = q0 + r + hi * 8;
            float a = s0[r] * SCALEQ, c = s1[r] * SCALEQ;
            int d0 = m - k0i, d1 = m - k1i;
            bool v0, v1;
            if (mode == 0)      { v0 = (d0 >= -WINSZ && d0 <= WINSZ);
                                  v1 = (d1 >= -WINSZ && d1 <= WINSZ); }
            else if (mode == 1) { v0 = (d0 >= 0 && d0 <= WINSZ);
                                  v1 = (d1 >= 0 && d1 <= WINSZ); }
            else                { v0 = v1 = true; }
            if (kp0) v0 = false;
            if (kp1) v1 = false;
            a = v0 ? a : NEGF;
            c = v1 ? c : NEGF;

            // row max across the 16 key-lanes of this half-wave
            float t = fmaxf(a, c);
            t = fmaxf(t, __shfl_xor(t, 1, 32));
            t = fmaxf(t, __shfl_xor(t, 2, 32));
            t = fmaxf(t, __shfl_xor(t, 4, 32));
            t = fmaxf(t, __shfl_xor(t, 8, 32));
            float mnew  = fmaxf(mrun[r], t);
            float alpha = __expf(mrun[r] - mnew);     // 0 on first valid tile
            float p0 = __expf(a - mnew);
            float p1 = __expf(c - mnew);
            float rs = p0 + p1;
            rs += __shfl_xor(rs, 1, 32);
            rs += __shfl_xor(rs, 2, 32);
            rs += __shfl_xor(rs, 4, 32);
            rs += __shfl_xor(rs, 8, 32);
            lrun[r] = lrun[r] * alpha + rs;
            mrun[r] = mnew;
#pragma unroll
            for (int j = 0; j < 4; ++j) oacc[j][r] *= alpha;

            P[(r + hi * 8) * 40 + nn]      = (_Float16)p0;
            P[(r + hi * 8) * 40 + 16 + nn] = (_Float16)p1;
        }

        // LDS round trip transposes C-layout P into A-layout fragment.
        asm volatile("s_wait_dscnt 0" ::: "memory");
        v16h pf = load_frag_a(&P[0], 40, 0, 0);

        // out += P(16x32) * V(32x64): 4 WMMAs over hd subtiles
#pragma unroll
        for (int j = 0; j < 4; ++j) {
            v16h vf = *(const v16h*)(Vh + (size_t)(16 * j + nn) * Lk + kt + hi * 16);
            oacc[j] = wmma_f16(pf, vf, oacc[j]);
        }
        asm volatile("s_wait_dscnt 0" ::: "memory"); // P reads done before next tile's writes
    }

#pragma unroll
    for (int r = 0; r < 8; ++r) {
        float inv = 1.0f / fmaxf(lrun[r], 1e-20f);
        int m = q0 + r + hi * 8;
#pragma unroll
        for (int j = 0; j < 4; ++j)
            O[((size_t)b * Lq + m) * DMODEL + h * HDIM + 16 * j + nn] =
                (_Float16)(oacc[j][r] * inv);
    }
}

// ---------------------------------------------------------------------------
// Support kernels
// ---------------------------------------------------------------------------
// fp32 W[K,N] -> f16 Wt[N,K]
__global__ void cvt_transpose(const float* __restrict__ W, _Float16* __restrict__ Wt,
                              int K, int N) {
    size_t i = (size_t)blockIdx.x * blockDim.x + threadIdx.x;
    if (i >= (size_t)K * N) return;
    int k = (int)(i / N), n = (int)(i % N);
    Wt[(size_t)n * K + k] = (_Float16)W[i];
}

// x = emb[tok]*sqrt(D) + sinusoidal PE, written f16
__global__ void embed_pe(const int* __restrict__ tok, const float* __restrict__ emb,
                         _Float16* __restrict__ X, int L) {
    size_t i = (size_t)blockIdx.x * blockDim.x + threadIdx.x;
    if (i >= (size_t)BB * L * DMODEL) return;
    int d = (int)(i % DMODEL);
    size_t row = i / DMODEL;
    int pos = (int)(row % L);
    int t = tok[row];
    float ang = (float)pos * __expf(-(float)(d & ~1) * (9.210340371976184f / DMODEL));
    float pe = (d & 1) ? __cosf(ang) : __sinf(ang);
    X[i] = (_Float16)(emb[(size_t)t * DMODEL + d] * SQRTD + pe);
}

// V [B*L, D] -> Vt [B, H, HD, L]  (contraction dim contiguous for P*V WMMA)
__global__ void transpose_heads(const _Float16* __restrict__ V, _Float16* __restrict__ Vt,
                                int L) {
    size_t i = (size_t)blockIdx.x * blockDim.x + threadIdx.x;
    if (i >= (size_t)BB * L * DMODEL) return;
    int c = (int)(i % DMODEL);
    size_t row = i / DMODEL;
    int l = (int)(row % L);
    int b = (int)(row / L);
    int h = c / HDIM, hd = c % HDIM;
    Vt[((size_t)(b * NHEAD + h) * HDIM + hd) * L + l] = V[i];
}

// out = LayerNorm(x + delta) * g + b   (one 256-thread block per row, D=512)
__global__ __launch_bounds__(256) void add_ln(
        const _Float16* __restrict__ X, const _Float16* __restrict__ Dl,
        const float* __restrict__ g, const float* __restrict__ bta,
        _Float16* __restrict__ Out) {
    __shared__ float s1[256], s2[256];
    size_t row = blockIdx.x;
    int t = threadIdx.x;
    float v0 = (float)X[row * DMODEL + t]       + (float)Dl[row * DMODEL + t];
    float v1 = (float)X[row * DMODEL + 256 + t] + (float)Dl[row * DMODEL + 256 + t];
    s1[t] = v0 + v1;
    s2[t] = v0 * v0 + v1 * v1;
    __syncthreads();
    for (int off = 128; off > 0; off >>= 1) {
        if (t < off) { s1[t] += s1[t + off]; s2[t] += s2[t + off]; }
        __syncthreads();
    }
    float mean = s1[0] * (1.0f / DMODEL);
    float var  = s2[0] * (1.0f / DMODEL) - mean * mean;
    float rstd = rsqrtf(var + 1e-5f);
    Out[row * DMODEL + t]       = (_Float16)((v0 - mean) * rstd * g[t] + bta[t]);
    Out[row * DMODEL + 256 + t] = (_Float16)((v1 - mean) * rstd * g[256 + t] + bta[256 + t]);
}

// ---------------------------------------------------------------------------
// Host orchestration
// ---------------------------------------------------------------------------
extern "C" void kernel_launch(void* const* d_in, const int* in_sizes, int n_in,
                              void* d_out, int out_size, void* d_ws, size_t ws_size,
                              hipStream_t stream) {
    (void)in_sizes; (void)n_in; (void)out_size; (void)ws_size;

    // --- flattened input order (setup_inputs dict order) ---
    // 0 src, 1 tgt, 2 src_kpm, 3 tgt_kpm, 4 src_emb, 5 tgt_emb,
    // enc layer i @ 6+16i : Wq Wk Wv Wo bq bk bv bo W1 b1 W2 b2 ln1g ln1b ln2g ln2b
    // dec layer i @ 70+26i: (same 16) caWq caWk caWv caWo cabq cabk cabv cabo ln3g ln3b
    // 174 Wg, 175 bg
    const int*  src = (const int*)d_in[0];
    const int*  tgt = (const int*)d_in[1];
    const unsigned char* skpm = (const unsigned char*)d_in[2];  // bool mask, 1 byte/elem
    const unsigned char* tkpm = (const unsigned char*)d_in[3];
    const float* src_emb = (const float*)d_in[4];
    const float* tgt_emb = (const float*)d_in[5];

    const int M = BB * SS;  // 4096 rows (== BB*TTLEN)

    // workspace carve-out
    size_t cur = 0;
    auto take = [&](size_t bytes) -> char* {
        char* r = (char*)d_ws + cur;
        cur += (bytes + 255) & ~(size_t)255;
        return r;
    };
    _Float16* xb  = (_Float16*)take((size_t)M * DMODEL * 2);   // encoder stream / memory
    _Float16* yb  = (_Float16*)take((size_t)M * DMODEL * 2);   // decoder stream
    _Float16* qb  = (_Float16*)take((size_t)M * DMODEL * 2);
    _Float16* kb  = (_Float16*)take((size_t)M * DMODEL * 2);
    _Float16* vb  = (_Float16*)take((size_t)M * DMODEL * 2);
    _Float16* vtb = (_Float16*)take((size_t)M * DMODEL * 2);
    _Float16* ab  = (_Float16*)take((size_t)M * DMODEL * 2);
    _Float16* tb  = (_Float16*)take((size_t)M * DMODEL * 2);
    _Float16* hb  = (_Float16*)take((size_t)M * FFDIM * 2);

    _Float16* ew[NENC][6];
    for (int li = 0; li < NENC; ++li) {
        for (int j = 0; j < 4; ++j) ew[li][j] = (_Float16*)take((size_t)DMODEL * DMODEL * 2);
        ew[li][4] = (_Float16*)take((size_t)DMODEL * FFDIM * 2);
        ew[li][5] = (_Float16*)take((size_t)FFDIM * DMODEL * 2);
    }
    _Float16* dw[NDEC][10];  // 0-3 sa WqWkWvWo, 4 W1, 5 W2, 6-9 ca WqWkWvWo
    for (int li = 0; li < NDEC; ++li) {
        for (int j = 0; j < 4; ++j) dw[li][j] = (_Float16*)take((size_t)DMODEL * DMODEL * 2);
        dw[li][4] = (_Float16*)take((size_t)DMODEL * FFDIM * 2);
        dw[li][5] = (_Float16*)take((size_t)FFDIM * DMODEL * 2);
        for (int j = 6; j < 10; ++j) dw[li][j] = (_Float16*)take((size_t)DMODEL * DMODEL * 2);
    }
    _Float16* wg = (_Float16*)take((size_t)VOCAB * DMODEL * 2);

    auto cvt = [&](const void* w, _Float16* wt, int K, int N) {
        size_t tot = (size_t)K * N;
        cvt_transpose<<<dim3((unsigned)((tot + 255) / 256)), 256, 0, stream>>>(
            (const float*)w, wt, K, N);
    };
    auto gemm = [&](const _Float16* A, const _Float16* Wt, const float* bias,
                    _Float16* C, int Mr, int Nc, int Kc, bool relu) {
        dim3 g(Nc / 128, Mr / 128);
        if (relu) gemm_wmma<_Float16, true ><<<g, 256, 0, stream>>>(A, Wt, bias, C, Mr, Nc, Kc);
        else      gemm_wmma<_Float16, false><<<g, 256, 0, stream>>>(A, Wt, bias, C, Mr, Nc, Kc);
    };
    auto elems = (size_t)M * DMODEL;
    auto egrid = dim3((unsigned)((elems + 255) / 256));

    // --- weight conversion (every call; deterministic) ---
    for (int li = 0; li < NENC; ++li) {
        int base = 6 + 16 * li;
        for (int j = 0; j < 4; ++j) cvt(d_in[base + j], ew[li][j], DMODEL, DMODEL);
        cvt(d_in[base + 8],  ew[li][4], DMODEL, FFDIM);
        cvt(d_in[base + 10], ew[li][5], FFDIM, DMODEL);
    }
    for (int li = 0; li < NDEC; ++li) {
        int base = 70 + 26 * li;
        for (int j = 0; j < 4; ++j) cvt(d_in[base + j], dw[li][j], DMODEL, DMODEL);
        cvt(d_in[base + 8],  dw[li][4], DMODEL, FFDIM);
        cvt(d_in[base + 10], dw[li][5], FFDIM, DMODEL);
        for (int j = 0; j < 4; ++j) cvt(d_in[base + 16 + j], dw[li][6 + j], DMODEL, DMODEL);
    }
    cvt(d_in[174], wg, DMODEL, VOCAB);

    // --- encoder ---
    embed_pe<<<egrid, 256, 0, stream>>>(src, src_emb, xb, SS);
    for (int li = 0; li < NENC; ++li) {
        int base = 6 + 16 * li;
        const float* bq = (const float*)d_in[base + 4];
        const float* bk = (const float*)d_in[base + 5];
        const float* bv = (const float*)d_in[base + 6];
        const float* bo = (const float*)d_in[base + 7];
        const float* b1 = (const float*)d_in[base + 9];
        const float* b2 = (const float*)d_in[base + 11];
        gemm(xb, ew[li][0], bq, qb, M, DMODEL, DMODEL, false);
        gemm(xb, ew[li][1], bk, kb, M, DMODEL, DMODEL, false);
        gemm(xb, ew[li][2], bv, vb, M, DMODEL, DMODEL, false);
        transpose_heads<<<egrid, 256, 0, stream>>>(vb, vtb, SS);
        attn_wmma<<<dim3(SS / 16, NHEAD, BB), 32, 0, stream>>>(
            qb, kb, vtb, skpm, ab, SS, SS, /*mode=*/0);
        gemm(ab, ew[li][3], bo, tb, M, DMODEL, DMODEL, false);
        add_ln<<<M, 256, 0, stream>>>(xb, tb,
            (const float*)d_in[base + 12], (const float*)d_in[base + 13], xb);
        gemm(xb, ew[li][4], b1, hb, M, FFDIM, DMODEL, true);
        gemm(hb, ew[li][5], b2, tb, M, DMODEL, FFDIM, false);
        add_ln<<<M, 256, 0, stream>>>(xb, tb,
            (const float*)d_in[base + 14], (const float*)d_in[base + 15], xb);
    }

    // --- decoder (xb is now the encoder memory) ---
    embed_pe<<<egrid, 256, 0, stream>>>(tgt, tgt_emb, yb, TTLEN);
    for (int li = 0; li < NDEC; ++li) {
        int base = 70 + 26 * li;
        const float* bq  = (const float*)d_in[base + 4];
        const float* bk  = (const float*)d_in[base + 5];
        const float* bv  = (const float*)d_in[base + 6];
        const float* bo  = (const float*)d_in[base + 7];
        const float* b1  = (const float*)d_in[base + 9];
        const float* b2  = (const float*)d_in[base + 11];
        const float* cbq = (const float*)d_in[base + 20];
        const float* cbk = (const float*)d_in[base + 21];
        const float* cbv = (const float*)d_in[base + 22];
        const float* cbo = (const float*)d_in[base + 23];
        // self-attention (causal window)
        gemm(yb, dw[li][0], bq, qb, M, DMODEL, DMODEL, false);
        gemm(yb, dw[li][1], bk, kb, M, DMODEL, DMODEL, false);
        gemm(yb, dw[li][2], bv, vb, M, DMODEL, DMODEL, false);
        transpose_heads<<<egrid, 256, 0, stream>>>(vb, vtb, TTLEN);
        attn_wmma<<<dim3(TTLEN / 16, NHEAD, BB), 32, 0, stream>>>(
            qb, kb, vtb, tkpm, ab, TTLEN, TTLEN, /*mode=*/1);
        gemm(ab, dw[li][3], bo, tb, M, DMODEL, DMODEL, false);
        add_ln<<<M, 256, 0, stream>>>(yb, tb,
            (const float*)d_in[base + 12], (const float*)d_in[base + 13], yb);
        // cross-attention (full, keys = encoder memory)
        gemm(yb, dw[li][6], cbq, qb, M, DMODEL, DMODEL, false);
        gemm(xb, dw[li][7], cbk, kb, M, DMODEL, DMODEL, false);
        gemm(xb, dw[li][8], cbv, vb, M, DMODEL, DMODEL, false);
        transpose_heads<<<egrid, 256, 0, stream>>>(vb, vtb, SS);
        attn_wmma<<<dim3(TTLEN / 16, NHEAD, BB), 32, 0, stream>>>(
            qb, kb, vtb, skpm, ab, TTLEN, SS, /*mode=*/2);
        gemm(ab, dw[li][9], cbo, tb, M, DMODEL, DMODEL, false);
        add_ln<<<M, 256, 0, stream>>>(yb, tb,
            (const float*)d_in[base + 14], (const float*)d_in[base + 15], yb);
        // FFN
        gemm(yb, dw[li][4], b1, hb, M, FFDIM, DMODEL, true);
        gemm(hb, dw[li][5], b2, tb, M, DMODEL, FFDIM, false);
        add_ln<<<M, 256, 0, stream>>>(yb, tb,
            (const float*)d_in[base + 24], (const float*)d_in[base + 25], yb);
    }

    // --- logits: [4096, 512] x [16000, 512]^T + bg -> fp32 d_out ---
    dim3 gg(VOCAB / 128, M / 128);
    gemm_wmma<float, false><<<gg, 256, 0, stream>>>(
        yb, wg, (const float*)d_in[175], (float*)d_out, M, VOCAB, DMODEL);
}
